// RelationGraphAttention_8735963480238
// MI455X (gfx1250) — matrix operand order repaired and checked by reading, not compiled, not run on hardware
//
#include <hip/hip_runtime.h>
#include <hip/hip_bf16.h>
#include <math.h>

// Problem constants (from reference)
#define BB   16
#define NN   256
#define CIN  256
#define HH   8
#define DD   64
#define HD   512           // H*D
#define E1   5             // E+1
#define HRW  3072          // HD + E1*HD
#define ROWS 4096          // B*N
#define HRCT (HRW / 16)    // 192 col tiles in hr/Wcat

typedef __attribute__((ext_vector_type(16))) _Float16 v16h;
typedef __attribute__((ext_vector_type(8)))  float    v8f;

// ---------------------------------------------------------------------------
// WMMA fragment layouts (V_WMMA_F32_16X16X32_F16, wave32), per ISA tables:
// A (16x32): lanes 0-15 M=lane, K in {0..7,16..23}; lanes 16-31 M=lane-16,
//            K in {8..15,24..31}; elem = (K&7) + 8*(K>>4).
// B (32x16): lane = n + 16*((K>>3)&1), elem = (K&7) + 8*(K>>4).
// C/D (16x16 f32): VGPR r -> M = r + 8*(lane>=16), N = lane&15.
//
// Swizzled storage: tiles stored as [tile][lane(32)][elem(16)] so a fragment
// is ONE contiguous 32-byte per-lane load.
// ---------------------------------------------------------------------------
__device__ __forceinline__ v16h load_frag_sw(const _Float16* base, size_t tile, int lane) {
  return *(const v16h*)(base + ((tile * 32 + (size_t)lane) << 4));
}

// Row-major A load (coalesces to 2x b128 per lane)
__device__ __forceinline__ v16h load_frag_a(const _Float16* A, int lda, int lane) {
  int m  = lane & 15;
  int kb = (lane >> 4) ? 8 : 0;
  v16h a;
#pragma unroll
  for (int v = 0; v < 8; ++v) {
    int k = ((v & 4) ? 16 : 0) + kb + ((v & 3) << 1);
    a[2 * v]     = A[(size_t)m * lda + k];
    a[2 * v + 1] = A[(size_t)m * lda + k + 1];
  }
  return a;
}

__device__ __forceinline__ v8f wmma_f16(v16h a, v16h b, v8f c) {
  return __builtin_amdgcn_wmma_f32_16x16x32_f16(false, a, false, b, (short)0, c,
                                                false, false);
}

// B-swizzle flat index for element (row=k, col) with nct col-tiles
__device__ __forceinline__ size_t bsw_idx(int row, int col, int nct) {
  int kt = row >> 5, kk = row & 31, ct = col >> 4, n = col & 15;
  int lane = n + (((kk >> 3) & 1) << 4);
  int elem = (kk & 7) + ((kk >> 4) << 3);
  return ((((size_t)kt * nct + ct) * 32 + lane) << 4) + elem;
}

// A-swizzle flat index for element (row, col=k) with nkt k-tiles
__device__ __forceinline__ size_t asw_idx(int row, int col, int nkt) {
  int rt = row >> 4, m = row & 15, kt = col >> 5, kk = col & 31;
  int lane = m + (((kk >> 3) & 1) << 4);
  int elem = (kk & 7) + ((kk >> 4) << 3);
  return ((((size_t)rt * nkt + kt) * 32 + lane) << 4) + elem;
}

__device__ __forceinline__ float gelu_exact(float x) {
  return 0.5f * x * (1.0f + erff(x * 0.70710678118654752f));
}

// ---------------------------------------------------------------------------
// K1a: concat W_hs|W_hr -> f16, B-fragment swizzled [256 x 3072]
// ---------------------------------------------------------------------------
__global__ __launch_bounds__(256) void k_cvt_wcat(const float* __restrict__ Whs,
                                                  const float* __restrict__ Whr,
                                                  _Float16* __restrict__ Wcat) {
  int idx = blockIdx.x * 256 + threadIdx.x;     // 786432 total
  int r = idx / HRW, c = idx % HRW;
  float v = (c < HD) ? Whs[r * HD + c] : Whr[r * (HRW - HD) + (c - HD)];
  Wcat[bsw_idx(r, c, HRCT)] = (_Float16)v;
}

// K1b: W_out -> f16, B-fragment swizzled [512 x 256]
__global__ __launch_bounds__(256) void k_cvt_wout(const float* __restrict__ W,
                                                  _Float16* __restrict__ Wo) {
  int idx = blockIdx.x * 256 + threadIdx.x;     // 131072 total
  int r = idx >> 8, c = idx & 255;
  Wo[bsw_idx(r, c, 16)] = (_Float16)W[idx];
}

// ---------------------------------------------------------------------------
// K2: LayerNorm over C_IN, write f16 xn [4096][256] row-major.
// ---------------------------------------------------------------------------
__global__ __launch_bounds__(256) void k_layernorm(const float* __restrict__ x,
                                                   const float* __restrict__ gamma,
                                                   const float* __restrict__ beta,
                                                   _Float16* __restrict__ xn) {
  __shared__ float red[256];
  int row = blockIdx.x, t = threadIdx.x;
  float v = x[(size_t)row * CIN + t];
  red[t] = v;
  __syncthreads();
  for (int s = 128; s > 0; s >>= 1) {
    if (t < s) red[t] += red[t + s];
    __syncthreads();
  }
  float mu = red[0] * (1.0f / CIN);
  __syncthreads();
  float d = v - mu;
  red[t] = d * d;
  __syncthreads();
  for (int s = 128; s > 0; s >>= 1) {
    if (t < s) red[t] += red[t + s];
    __syncthreads();
  }
  float var = red[0] * (1.0f / CIN);
  float rs  = rsqrtf(var + 1e-5f);
  xn[(size_t)row * CIN + t] = (_Float16)(d * rs * gamma[t] + beta[t]);
}

// ---------------------------------------------------------------------------
// K3: WMMA GEMM  hr = xn @ Wcat + bias.  Output stored B-swizzled for K6.
// Wave computes 16 rows x 32 cols (2 tiles, A reused). Block = 8 waves = 256c.
// grid = (3072/256=12, 4096/16=256)
// ---------------------------------------------------------------------------
__global__ __launch_bounds__(256) void k_gemm_hr(const _Float16* __restrict__ xn,
                                                 const _Float16* __restrict__ Wcat,
                                                 const float* __restrict__ bhs,
                                                 const float* __restrict__ bhr,
                                                 _Float16* __restrict__ hr) {
  int wave = threadIdx.x >> 5, lane = threadIdx.x & 31;
  int row0 = blockIdx.y * 16;
  int ct0  = blockIdx.x * 16 + wave * 2;     // first of two col tiles
  v8f acc0 = {0.f, 0.f, 0.f, 0.f, 0.f, 0.f, 0.f, 0.f};
  v8f acc1 = acc0;
#pragma unroll
  for (int kt = 0; kt < CIN / 32; ++kt) {
    v16h a  = load_frag_a(xn + (size_t)row0 * CIN + kt * 32, CIN, lane);
    v16h b0 = load_frag_sw(Wcat, (size_t)kt * HRCT + ct0, lane);
    v16h b1 = load_frag_sw(Wcat, (size_t)kt * HRCT + ct0 + 1, lane);
    acc0 = wmma_f16(a, b0, acc0);
    acc1 = wmma_f16(a, b1, acc1);
  }
  int n  = lane & 15;
  int mb = (lane >> 4) ? 8 : 0;
#pragma unroll
  for (int tile = 0; tile < 2; ++tile) {
    int col = (ct0 + tile) * 16 + n;
    float bias = (col < HD) ? bhs[col] : bhr[col - HD];
    const v8f& acc = tile ? acc1 : acc0;
#pragma unroll
    for (int r = 0; r < 8; ++r) {
      int row = row0 + mb + r;
      hr[bsw_idx(row, col, HRCT)] = (_Float16)(acc[r] + bias);
    }
  }
}

// ---------------------------------------------------------------------------
// K4: attention dots from B-swizzled hr. hs_attn[4096][8], hr_attn[4096][5][8]
// ---------------------------------------------------------------------------
__global__ __launch_bounds__(64) void k_dots(const _Float16* __restrict__ hr,
                                             const float* __restrict__ aw,
                                             float* __restrict__ hs_attn,
                                             float* __restrict__ hr_attn) {
  int row = blockIdx.x, q = threadIdx.x;
  if (q >= 48) return;
  if (q < 8) {
    int h = q;
    float s = 0.f;
    for (int d = 0; d < DD; ++d)
      s += (float)hr[bsw_idx(row, h * DD + d, HRCT)] * aw[h * 2 * DD + d];
    hs_attn[row * HH + h] = s;
  } else {
    int e = (q - 8) >> 3, h = (q - 8) & 7;
    float s = 0.f;
    for (int d = 0; d < DD; ++d)
      s += (float)hr[bsw_idx(row, HD + e * HD + h * DD + d, HRCT)] *
           aw[h * 2 * DD + DD + d];
    hr_attn[(row * E1 + e) * HH + h] = s;
  }
}

// ---------------------------------------------------------------------------
// K5: compress one-hot adjacency -> e_sel int8 (-1 none, 0..3 edge, 4 self)
// ---------------------------------------------------------------------------
__global__ __launch_bounds__(256) void k_esel(const float* __restrict__ adj,
                                              signed char* __restrict__ esel) {
  size_t idx = (size_t)blockIdx.x * 256 + threadIdx.x;  // B*N*N = 1048576
  int j = (int)(idx & 255), i = (int)((idx >> 8) & 255);
  signed char e = -1;
  if (i == j) {
    e = 4;
  } else {
    const float* a = adj + idx * 4;
#pragma unroll
    for (int q = 0; q < 4; ++q)
      if (a[q] > 0.5f) e = (signed char)q;
  }
  esel[idx] = e;
}

// ---------------------------------------------------------------------------
// K6: fused attention. One block per (b, 16-row i-tile).  grid = 256.
// Softmax rows resident in LDS; per-head wave builds masked-probability A
// fragments in registers, streams B fragments (single 32B vector loads from
// swizzled hr), skips absent edge types via ballot. GELU epilogue writes z
// in A-fragment order for K7.
// ---------------------------------------------------------------------------
#define LJ 257
__global__ __launch_bounds__(256) void k_attn(const _Float16* __restrict__ hr,
                                              const float* __restrict__ hs_attn,
                                              const float* __restrict__ hr_attn,
                                              const signed char* __restrict__ esel,
                                              _Float16* __restrict__ z) {
  __shared__ _Float16 sL[16][HH][LJ];        // logits, f16
  __shared__ float sHr[NN][E1 * HH];         // hr_attn_node for all j
  __shared__ signed char sE[16][NN];
  __shared__ float sHs[16][HH];
  __shared__ float sMax[16][HH];
  __shared__ float sInv[16][HH];

  int t = threadIdx.x;
  int b  = blockIdx.x >> 4;
  int i0 = (blockIdx.x & 15) * 16;
  int rowbase = b * NN;

  for (int idx = t; idx < NN * E1 * HH; idx += 256)
    sHr[idx / (E1 * HH)][idx % (E1 * HH)] = hr_attn[(size_t)rowbase * E1 * HH + idx];
  for (int idx = t; idx < 16 * NN; idx += 256)
    sE[idx >> 8][idx & 255] =
        esel[(size_t)b * NN * NN + (size_t)(i0 + (idx >> 8)) * NN + (idx & 255)];
  if (t < 128) sHs[t >> 3][t & 7] = hs_attn[(size_t)(rowbase + i0 + (t >> 3)) * HH + (t & 7)];
  __syncthreads();

  // Phase 1: logits
  for (int rep = 0; rep < 16; ++rep) {
    int p = rep * 256 + t;
    int i = p >> 8, j = p & 255;
    signed char e = sE[i][j];
#pragma unroll
    for (int h = 0; h < HH; ++h) {
      float l;
      if (e >= 0) {
        l = sHs[i][h] + sHr[j][e * HH + h];
        l = (l > 0.f) ? l : 0.2f * l;      // leaky_relu(0.2)
      } else {
        l = -60000.f;                       // mask (f16-safe)
      }
      sL[i][h][j] = (_Float16)l;
    }
  }
  __syncthreads();

  // Phase 1b: per-(i,h) max and 1/sum
  if (t < 128) {
    int i = t >> 3, h = t & 7;
    float mx = -1e30f;
    for (int j = 0; j < NN; ++j) {
      float l = (float)sL[i][h][j];
      mx = l > mx ? l : mx;
    }
    float s = 0.f;
    for (int j = 0; j < NN; ++j) s += __expf((float)sL[i][h][j] - mx);
    sMax[i][h] = mx;
    sInv[i][h] = 1.f / s;
  }
  __syncthreads();

  // Phase 2: wave w handles head h = w. 4 accumulators cover D=64.
  int wave = t >> 5, lane = t & 31;
  int h  = wave;
  int m  = lane & 15;
  int kb = (lane >> 4) ? 8 : 0;
  float rmax = sMax[m][h], rinv = sInv[m][h];
  v8f acc0 = {0.f,0.f,0.f,0.f,0.f,0.f,0.f,0.f};
  v8f acc1 = acc0, acc2 = acc0, acc3 = acc0;

  for (int j0 = 0; j0 < NN; j0 += 32) {
    _Float16 pv[16];
    signed char ev[16];
    unsigned emask = 0;
#pragma unroll
    for (int v = 0; v < 8; ++v) {
      int k = ((v & 4) ? 16 : 0) + kb + ((v & 3) << 1);
#pragma unroll
      for (int u = 0; u < 2; ++u) {
        int j = j0 + k + u;
        signed char e = sE[m][j];
        float l  = (float)sL[m][h][j];
        float pp = __expf(l - rmax) * rinv;
        pv[2 * v + u] = (_Float16)pp;
        ev[2 * v + u] = e;
        if (e >= 0) emask |= (1u << e);
      }
    }
    size_t jt = (size_t)(rowbase + j0) >> 5;
#pragma unroll
    for (int e = 0; e < E1; ++e) {
      if (__ballot((int)((emask >> e) & 1u)) == 0ull) continue;   // no such edges in tile
      v16h a;
#pragma unroll
      for (int q = 0; q < 16; ++q) a[q] = (ev[q] == e) ? pv[q] : (_Float16)0.f;
      size_t ctb = 32 + e * 32 + h * 4;    // col-tile base: (HD + e*HD + h*DD)/16
      acc0 = wmma_f16(a, load_frag_sw(hr, jt * HRCT + ctb + 0, lane), acc0);
      acc1 = wmma_f16(a, load_frag_sw(hr, jt * HRCT + ctb + 1, lane), acc1);
      acc2 = wmma_f16(a, load_frag_sw(hr, jt * HRCT + ctb + 2, lane), acc2);
      acc3 = wmma_f16(a, load_frag_sw(hr, jt * HRCT + ctb + 3, lane), acc3);
    }
  }

  // Epilogue: GELU -> z, stored A-fragment swizzled ([4096 x 512], 16 k-tiles)
  int n  = lane & 15;
  int mb = (lane >> 4) ? 8 : 0;
#pragma unroll
  for (int r = 0; r < 8; ++r) {
    int row = rowbase + i0 + mb + r;
    z[asw_idx(row, h * DD +  0 + n, 16)] = (_Float16)gelu_exact(acc0[r]);
    z[asw_idx(row, h * DD + 16 + n, 16)] = (_Float16)gelu_exact(acc1[r]);
    z[asw_idx(row, h * DD + 32 + n, 16)] = (_Float16)gelu_exact(acc2[r]);
    z[asw_idx(row, h * DD + 48 + n, 16)] = (_Float16)gelu_exact(acc3[r]);
  }
}

// ---------------------------------------------------------------------------
// K7: WMMA output GEMM  out[4096][256] = gelu(attn) @ Wout + b_out (f32).
// A from swizzled z, B from swizzled Wout: all fragment loads contiguous.
// Wave does 2 col tiles; block covers all 256 cols. grid = (1, 256).
// ---------------------------------------------------------------------------
__global__ __launch_bounds__(256) void k_gemm_out(const _Float16* __restrict__ zf,
                                                  const _Float16* __restrict__ Wo,
                                                  const float* __restrict__ bout,
                                                  float* __restrict__ out) {
  int wave = threadIdx.x >> 5, lane = threadIdx.x & 31;
  int rt   = blockIdx.y;
  int row0 = rt * 16;
  int ct0  = wave * 2;
  v8f acc0 = {0.f, 0.f, 0.f, 0.f, 0.f, 0.f, 0.f, 0.f};
  v8f acc1 = acc0;
#pragma unroll
  for (int kt = 0; kt < HD / 32; ++kt) {
    v16h a  = load_frag_sw(zf, (size_t)rt * 16 + kt, lane);
    v16h b0 = load_frag_sw(Wo, (size_t)kt * 16 + ct0, lane);
    v16h b1 = load_frag_sw(Wo, (size_t)kt * 16 + ct0 + 1, lane);
    acc0 = wmma_f16(a, b0, acc0);
    acc1 = wmma_f16(a, b1, acc1);
  }
  int n  = lane & 15;
  int mb = (lane >> 4) ? 8 : 0;
#pragma unroll
  for (int tile = 0; tile < 2; ++tile) {
    int col = (ct0 + tile) * 16 + n;
    float bias = bout[col];
    const v8f& acc = tile ? acc1 : acc0;
#pragma unroll
    for (int r = 0; r < 8; ++r) {
      int row = row0 + mb + r;
      out[(size_t)row * 256 + col] = acc[r] + bias;
    }
  }
}

// ---------------------------------------------------------------------------
extern "C" void kernel_launch(void* const* d_in, const int* in_sizes, int n_in,
                              void* d_out, int out_size, void* d_ws, size_t ws_size,
                              hipStream_t stream) {
  const float* x     = (const float*)d_in[0];
  const float* adj   = (const float*)d_in[1];
  const float* gamma = (const float*)d_in[2];
  const float* beta  = (const float*)d_in[3];
  const float* Whs   = (const float*)d_in[4];
  const float* bhs   = (const float*)d_in[5];
  const float* Whr   = (const float*)d_in[6];
  const float* bhr   = (const float*)d_in[7];
  const float* aw    = (const float*)d_in[8];
  const float* Wout  = (const float*)d_in[9];
  const float* bout  = (const float*)d_in[10];
  float* out = (float*)d_out;
  (void)in_sizes; (void)n_in; (void)out_size; (void)ws_size;

  char* ws = (char*)d_ws;
  size_t off = 0;
  auto alloc = [&](size_t bytes) -> void* {
    void* p = (void*)(ws + off);
    off = (off + bytes + 255) & ~(size_t)255;
    return p;
  };
  _Float16*    Wcat    = (_Float16*)alloc((size_t)CIN * HRW * 2);       // 1.5 MB
  _Float16*    Wo16    = (_Float16*)alloc((size_t)HD * 256 * 2);        // 256 KB
  _Float16*    xn      = (_Float16*)alloc((size_t)ROWS * CIN * 2);      // 2 MB
  _Float16*    hr      = (_Float16*)alloc((size_t)ROWS * HRW * 2);      // 24 MB
  float*       hs_attn = (float*)alloc((size_t)ROWS * HH * 4);
  float*       hr_attn = (float*)alloc((size_t)ROWS * E1 * HH * 4);
  signed char* esel    = (signed char*)alloc((size_t)BB * NN * NN);
  _Float16*    zf      = (_Float16*)alloc((size_t)ROWS * HD * 2);       // 4 MB

  k_cvt_wcat<<<(CIN * HRW) / 256, 256, 0, stream>>>(Whs, Whr, Wcat);
  k_cvt_wout<<<(HD * 256) / 256, 256, 0, stream>>>(Wout, Wo16);
  k_layernorm<<<ROWS, 256, 0, stream>>>(x, gamma, beta, xn);
  k_gemm_hr<<<dim3(HRW / 256, ROWS / 16), 256, 0, stream>>>(xn, Wcat, bhs, bhr, hr);
  k_dots<<<ROWS, 64, 0, stream>>>(hr, aw, hs_attn, hr_attn);
  k_esel<<<(BB * NN * NN) / 256, 256, 0, stream>>>(adj, esel);
  k_attn<<<BB * (NN / 16), 256, 0, stream>>>(hr, hs_attn, hr_attn, esel, zf);
  k_gemm_out<<<dim3(1, ROWS / 16), 256, 0, stream>>>(zf, Wo16, bout, out);
}
